// DeepNCM_24610162606421
// MI455X (gfx1250) — matrix operand order repaired and checked by reading, not compiled, not run on hardware
//
#include <hip/hip_runtime.h>
#include <hip/hip_bf16.h>

#define NCLS 1024
#define DIM  1024
#define NSMP 32768

typedef __attribute__((ext_vector_type(16))) __bf16 v16bf;
typedef __attribute__((ext_vector_type(8)))  float  v8f;

__device__ __forceinline__ unsigned short f32_to_bf16_rne(float f) {
    unsigned int u = __float_as_uint(f);
    unsigned int r = u + 0x7FFFu + ((u >> 16) & 1u);
    return (unsigned short)(r >> 16);
}
__device__ __forceinline__ float bf16_bits_to_f32(unsigned short h) {
    return __uint_as_float(((unsigned int)h) << 16);
}
__device__ __forceinline__ void split_bf16(float x, unsigned short& hi, unsigned short& lo) {
    hi = f32_to_bf16_rne(x);
    lo = f32_to_bf16_rne(x - bf16_bits_to_f32(hi));
}

// CDNA5 async DMA: global -> LDS, 16B per lane, tracked by ASYNCcnt.
__device__ __forceinline__ void async_load_b128(unsigned lds_off, const void* gptr) {
    asm volatile("global_load_async_to_lds_b128 %0, %1, off"
                 :: "v"(lds_off), "v"((unsigned long long)(size_t)gptr)
                 : "memory");
}
__device__ __forceinline__ void wait_async0() {
    asm volatile("s_wait_asynccnt 0x0" ::: "memory");
}
__device__ __forceinline__ unsigned lds_off_of(const void* p) {
    // generic LDS pointer: addr[31:0] == LDS byte offset (ISA 10.2 aperture map)
    return (unsigned)(unsigned long long)(size_t)p;
}

// ---------------------------------------------------------------------------
// K1: deterministic per-class scatter-mean + prototype blend + bf16 split.
// One block (256 threads) per class; thread t owns dims [4t, 4t+4).
// ---------------------------------------------------------------------------
__global__ __launch_bounds__(256) void proto_update_kernel(
    const float* __restrict__ E, const float* __restrict__ P,
    const float* __restrict__ counter, const int* __restrict__ y,
    unsigned short* __restrict__ Phi, unsigned short* __restrict__ Plo,
    float* __restrict__ p_sq)
{
    const int c    = blockIdx.x;
    const int t    = threadIdx.x;
    const int lane = t & 31;
    const int wv   = t >> 5;

    __shared__ int   s_list[2048];
    __shared__ int   s_wcnt[8];
    __shared__ float s_red[256];

    float a0 = 0.f, a1 = 0.f, a2 = 0.f, a3 = 0.f;
    int cnt_total = 0;
    const int d0 = t * 4;

    for (int chunk = 0; chunk < NSMP; chunk += 2048) {
        int nlist = 0;
        for (int pass = 0; pass < 8; ++pass) {
            const int s = chunk + pass * 256 + t;
            const bool m = (y[s] == c);
            const unsigned int mask = __builtin_amdgcn_ballot_w32(m);
            if (lane == 0) s_wcnt[wv] = __popc(mask);
            __syncthreads();
            int off = 0;
            for (int w = 0; w < wv; ++w) off += s_wcnt[w];
            const int prefix = __popc(mask & ((1u << lane) - 1u));
            if (m) s_list[nlist + off + prefix] = s;
            int tot = 0;
            for (int w = 0; w < 8; ++w) tot += s_wcnt[w];
            nlist += tot;
            __syncthreads();
        }
        // fixed-order accumulation -> deterministic fp result
        for (int m = 0; m < nlist; ++m) {
            const int s = s_list[m];
            const float4 e = *(const float4*)(E + (size_t)s * DIM + d0);
            a0 += e.x; a1 += e.y; a2 += e.z; a3 += e.w;
        }
        cnt_total += nlist;
        __syncthreads();
    }

    const float cnt   = (float)cnt_total;
    const float ctr   = counter[c];
    const float inv   = 1.0f / fmaxf(ctr + cnt, 1.0f);
    const bool  has   = cnt_total > 0;
    const float4 pv   = *(const float4*)(P + (size_t)c * DIM + d0);

    const float np0 = has ? (ctr * pv.x + a0) * inv : pv.x;
    const float np1 = has ? (ctr * pv.y + a1) * inv : pv.y;
    const float np2 = has ? (ctr * pv.z + a2) * inv : pv.z;
    const float np3 = has ? (ctr * pv.w + a3) * inv : pv.w;

    unsigned short h0, l0, h1, l1, h2, l2, h3, l3;
    split_bf16(np0, h0, l0); split_bf16(np1, h1, l1);
    split_bf16(np2, h2, l2); split_bf16(np3, h3, l3);

    uint2 hv, lv;
    hv.x = (unsigned)h0 | ((unsigned)h1 << 16); hv.y = (unsigned)h2 | ((unsigned)h3 << 16);
    lv.x = (unsigned)l0 | ((unsigned)l1 << 16); lv.y = (unsigned)l2 | ((unsigned)l3 << 16);
    *(uint2*)(Phi + (size_t)c * DIM + d0) = hv;
    *(uint2*)(Plo + (size_t)c * DIM + d0) = lv;

    s_red[t] = np0 * np0 + np1 * np1 + np2 * np2 + np3 * np3;
    __syncthreads();
    for (int s = 128; s > 0; s >>= 1) {
        if (t < s) s_red[t] += s_red[t + s];
        __syncthreads();
    }
    if (t == 0) p_sq[c] = s_red[0];
}

// ---------------------------------------------------------------------------
// K2: per-row ||e||^2. One block per row.
// ---------------------------------------------------------------------------
__global__ __launch_bounds__(256) void esq_kernel(const float* __restrict__ E,
                                                  float* __restrict__ e_sq)
{
    const int row = blockIdx.x;
    const int t = threadIdx.x;
    __shared__ float s_red[256];
    const float4 v = *(const float4*)(E + (size_t)row * DIM + t * 4);
    s_red[t] = v.x * v.x + v.y * v.y + v.z * v.z + v.w * v.w;
    __syncthreads();
    for (int s = 128; s > 0; s >>= 1) {
        if (t < s) s_red[t] += s_red[t + s];
        __syncthreads();
    }
    if (t == 0) e_sq[row] = s_red[0];
}

// ---------------------------------------------------------------------------
// K3: 128x128-tile GEMM via split-bf16 WMMA + fused -sqdist epilogue.
// 8 waves per block: 4x2 grid, each wave owns 2x4 tiles of 16x16.
// A tile: fp32 global -> VGPR split -> LDS. B tiles: async DMA global -> LDS.
// ---------------------------------------------------------------------------
#define BM 128
#define BN 128
#define BK 32
#define LPAD 8  // ushort padding -> 80B row stride, conflict-free frag reads

__global__ __launch_bounds__(256) void ncm_gemm_kernel(
    const float* __restrict__ E,
    const unsigned short* __restrict__ Phi, const unsigned short* __restrict__ Plo,
    const float* __restrict__ e_sq, const float* __restrict__ p_sq,
    float* __restrict__ out)
{
    __shared__ unsigned short sAhi[BM][BK + LPAD];
    __shared__ unsigned short sAlo[BM][BK + LPAD];
    __shared__ unsigned short sBhi[BN][BK + LPAD];
    __shared__ unsigned short sBlo[BN][BK + LPAD];

    const int t    = threadIdx.x;
    const int lane = t & 31;
    const int wv   = t >> 5;
    const int wr   = wv >> 1;      // 0..3 : row group (32 rows)
    const int wc   = wv & 1;       // 0..1 : col group (64 cols)
    const int half = lane >> 4;    // WMMA lane half
    const int lm   = lane & 15;
    const int Mb   = blockIdx.x * BM;
    const int Nb   = blockIdx.y * BN;

    typedef union { v16bf v; unsigned int u[8]; } frag16;
    typedef union { v8f  v; float f[8]; } frag8;

    frag8 acc[2][4];
    for (int rt = 0; rt < 2; ++rt)
        for (int ct = 0; ct < 4; ++ct)
            for (int i = 0; i < 8; ++i) acc[rt][ct].f[i] = 0.0f;

    const int q  = t & 7;   // E tile: 8 threads per row (float4 each)
    const int r0 = t >> 3;  // 0..31
    const int hB = t & 1;   // P tile: 2 threads per row, 32B each
    const int rB = t >> 1;  // 0..127

    for (int k0 = 0; k0 < DIM; k0 += BK) {
        // ---- B tiles: async DMA straight into LDS (ASYNCcnt path) ----
        {
            const size_t off = (size_t)(Nb + rB) * DIM + k0 + hB * 16;
            async_load_b128(lds_off_of(&sBhi[rB][hB * 16 + 0]), Phi + off + 0);
            async_load_b128(lds_off_of(&sBhi[rB][hB * 16 + 8]), Phi + off + 8);
            async_load_b128(lds_off_of(&sBlo[rB][hB * 16 + 0]), Plo + off + 0);
            async_load_b128(lds_off_of(&sBlo[rB][hB * 16 + 8]), Plo + off + 8);
        }
        // ---- A tile: fp32 -> bf16 hi/lo through VGPRs ----
        for (int p = 0; p < 4; ++p) {
            const int r = r0 + p * 32;
            const float* src = E + (size_t)(Mb + r) * DIM + k0 + q * 4;
            const float4 ev = *(const float4*)src;
            if (k0 + BK < DIM) __builtin_prefetch(src + BK, 0, 1);
            unsigned short hx, lx, hy, ly, hz, lz, hw, lw;
            split_bf16(ev.x, hx, lx); split_bf16(ev.y, hy, ly);
            split_bf16(ev.z, hz, lz); split_bf16(ev.w, hw, lw);
            uint2 hv, lv;
            hv.x = (unsigned)hx | ((unsigned)hy << 16); hv.y = (unsigned)hz | ((unsigned)hw << 16);
            lv.x = (unsigned)lx | ((unsigned)ly << 16); lv.y = (unsigned)lz | ((unsigned)lw << 16);
            *(uint2*)(&sAhi[r][q * 4]) = hv;
            *(uint2*)(&sAlo[r][q * 4]) = lv;
        }
        wait_async0();       // this wave's DMA writes landed in LDS
        __syncthreads();     // all waves' writes visible

        // ---- build fragments per ISA 7.12.2 layouts ----
        frag16 ahi[2], alo[2], bhi[4], blo[4];
        for (int rt = 0; rt < 2; ++rt) {
            const int mrow = wr * 32 + rt * 16 + lm;
            #pragma unroll
            for (int v = 0; v < 8; ++v) {
                const int k = ((v >> 2) << 4) + (half << 3) + ((v & 3) << 1);
                ahi[rt].u[v] = *(const unsigned int*)(&sAhi[mrow][k]);
                alo[rt].u[v] = *(const unsigned int*)(&sAlo[mrow][k]);
            }
        }
        for (int ct = 0; ct < 4; ++ct) {
            const int nrow = wc * 64 + ct * 16 + lm;
            #pragma unroll
            for (int v = 0; v < 8; ++v) {
                const int k = ((v >> 2) << 4) + (half << 3) + ((v & 3) << 1);
                bhi[ct].u[v] = *(const unsigned int*)(&sBhi[nrow][k]);
                blo[ct].u[v] = *(const unsigned int*)(&sBlo[nrow][k]);
            }
        }

        // ---- split-bf16 accumulation: hi*hi + hi*lo + lo*hi ----
        #pragma unroll
        for (int rt = 0; rt < 2; ++rt)
            #pragma unroll
            for (int ct = 0; ct < 4; ++ct) {
                acc[rt][ct].v = __builtin_amdgcn_wmma_f32_16x16x32_bf16(
                    false, ahi[rt].v, false, bhi[ct].v, (short)0, acc[rt][ct].v, false, false);
                acc[rt][ct].v = __builtin_amdgcn_wmma_f32_16x16x32_bf16(
                    false, ahi[rt].v, false, blo[ct].v, (short)0, acc[rt][ct].v, false, false);
                acc[rt][ct].v = __builtin_amdgcn_wmma_f32_16x16x32_bf16(
                    false, alo[rt].v, false, bhi[ct].v, (short)0, acc[rt][ct].v, false, false);
            }
        __syncthreads();
    }

    // ---- epilogue: out[m][n] = -max(e_sq[m] + p_sq[n] - 2*dot, 0) ----
    for (int rt = 0; rt < 2; ++rt) {
        float es[8];
        #pragma unroll
        for (int i = 0; i < 8; ++i)
            es[i] = e_sq[Mb + wr * 32 + rt * 16 + (half << 3) + i];
        for (int ct = 0; ct < 4; ++ct) {
            const int ncol = Nb + wc * 64 + ct * 16 + lm;
            const float psq = p_sq[ncol];
            #pragma unroll
            for (int i = 0; i < 8; ++i) {
                const int mrow = Mb + wr * 32 + rt * 16 + (half << 3) + i;
                const float d = es[i] + psq - 2.0f * acc[rt][ct].f[i];
                out[(size_t)mrow * NCLS + ncol] = -fmaxf(d, 0.0f);
            }
        }
    }
}

// ---------------------------------------------------------------------------
extern "C" void kernel_launch(void* const* d_in, const int* in_sizes, int n_in,
                              void* d_out, int out_size, void* d_ws, size_t ws_size,
                              hipStream_t stream) {
    const float* E       = (const float*)d_in[0];   // [32768,1024] f32
    const float* P       = (const float*)d_in[1];   // [1024,1024]  f32
    const float* counter = (const float*)d_in[2];   // [1024]       f32
    const int*   y       = (const int*)d_in[3];     // [32768]      i32
    float* out = (float*)d_out;                     // [32768,1024] f32

    char* ws = (char*)d_ws;
    unsigned short* Phi  = (unsigned short*)(ws);                        // 2 MB
    unsigned short* Plo  = (unsigned short*)(ws + ((size_t)2 << 20));    // 2 MB
    float*          p_sq = (float*)(ws + ((size_t)4 << 20));             // 4 KB
    float*          e_sq = (float*)(ws + ((size_t)4 << 20) + 4096);      // 128 KB

    proto_update_kernel<<<NCLS, 256, 0, stream>>>(E, P, counter, y, Phi, Plo, p_sq);
    esq_kernel<<<NSMP, 256, 0, stream>>>(E, e_sq);
    dim3 grid(NSMP / BM, NCLS / BN); // 256 x 8
    ncm_gemm_kernel<<<grid, 256, 0, stream>>>(E, Phi, Plo, e_sq, p_sq, out);
}
